// LigerFusedLinearPackDPOLoss_71975061946951
// MI455X (gfx1250) — matrix operand order repaired and checked by reading, not compiled, not run on hardware
//
#include <hip/hip_runtime.h>
#include <math.h>

typedef __attribute__((ext_vector_type(16))) __bf16 v16bf;
typedef __attribute__((ext_vector_type(8)))  __bf16 v8bf;
typedef __attribute__((ext_vector_type(8)))  float  v8f;

#define HD        2048
#define VOC       32000
#define ROWS      1024          /* B*S per stream */
#define NSTREAMS  4
#define MTILE     32            /* rows staged per workgroup */
#define ROWTILES  32            /* ROWS / MTILE */
#define NWAVES    8             /* 256 threads */
#define NCHUNK    1280          /* vocab columns per workgroup */
#define NCHUNKS   25            /* VOC / NCHUNK */
#define NITERS    5             /* NCHUNK / (NWAVES*32) */
#define IGNORE_IDX (-100)
#define BETA_C    0.1f
#define NEG_INF   (-3.402823466e38f)

__device__ __forceinline__ unsigned short f32_to_bf16_bits(float f) {
    union { float f; unsigned u; } v; v.f = f;
    unsigned r = v.u + 0x7FFFu + ((v.u >> 16) & 1u);   // round-to-nearest-even
    return (unsigned short)(r >> 16);
}
__device__ __forceinline__ __bf16 f32_to_bf16(float f) {
    unsigned short h = f32_to_bf16_bits(f);
    __bf16 out;
    __builtin_memcpy(&out, &h, 2);
    return out;
}

// Fold a 16x16 logit tile (v8f, C layout: row = v + 8*half, col spans a
// 16-lane group) into the per-row running (max, sumexp).
__device__ __forceinline__ void fold_tile(const v8f& c, float bn,
                                          float m_run[8], float s_run[8]) {
#pragma unroll
    for (int v = 0; v < 8; ++v) {
        float val = c[v] + bn;
        float tm = val;
        tm = fmaxf(tm, __shfl_xor(tm, 1, 16));
        tm = fmaxf(tm, __shfl_xor(tm, 2, 16));
        tm = fmaxf(tm, __shfl_xor(tm, 4, 16));
        tm = fmaxf(tm, __shfl_xor(tm, 8, 16));
        float te = __expf(val - tm);
        te += __shfl_xor(te, 1, 16);
        te += __shfl_xor(te, 2, 16);
        te += __shfl_xor(te, 4, 16);
        te += __shfl_xor(te, 8, 16);
        float nm = fmaxf(m_run[v], tm);
        s_run[v] = s_run[v] * __expf(m_run[v] - nm) + te * __expf(tm - nm);
        m_run[v] = nm;
    }
}

// ---------------------------------------------------------------------------
// Kernel 1: f32 -> bf16 weight conversion (policy + ref, back to back in ws)
// ---------------------------------------------------------------------------
__global__ void dpo_convert_w_kernel(const float* __restrict__ wp,
                                     const float* __restrict__ wr,
                                     unsigned short* __restrict__ out) {
    const size_t welts = (size_t)VOC * HD;
    size_t i4 = ((size_t)blockIdx.x * blockDim.x + threadIdx.x) * 4;
    if (i4 >= 2 * welts) return;
    const float* src = (i4 < welts) ? (wp + i4) : (wr + (i4 - welts));
    float4 q = *(const float4*)src;
    ushort4 o;
    o.x = f32_to_bf16_bits(q.x);
    o.y = f32_to_bf16_bits(q.y);
    o.z = f32_to_bf16_bits(q.z);
    o.w = f32_to_bf16_bits(q.w);
    *(ushort4*)(out + i4) = o;
}

// ---------------------------------------------------------------------------
// Kernel 2: fused linear -> per-token partial logsumexp over a vocab chunk.
// 256 threads (8 waves); 32-row x tile staged in dynamic LDS (128 KB) as bf16.
// Each wave register-blocks a 32x32 C region (2x2 WMMA tiles): per k-step
// 2 A frags (ds) + 2 B frags (global) feed 4 v_wmma_f32_16x16x32_bf16.
// ---------------------------------------------------------------------------
template <bool BF16W>
__global__ __launch_bounds__(256)
void dpo_gemm_lse_kernel(const float* __restrict__ x0, const float* __restrict__ x1,
                         const float* __restrict__ x2, const float* __restrict__ x3,
                         const void* __restrict__ wpol, const void* __restrict__ wref,
                         const float* __restrict__ bpol, const float* __restrict__ bref,
                         float* __restrict__ pm, float* __restrict__ ps) {
    const int bx    = blockIdx.x;
    const int s     = bx / (ROWTILES * NCHUNKS);
    const int rem   = bx % (ROWTILES * NCHUNKS);
    const int rt    = rem / NCHUNKS;
    const int chunk = rem % NCHUNKS;

    const float* xs = (s == 0) ? x0 : (s == 1) ? x1 : (s == 2) ? x2 : x3;
    const void*  W  = (s < 2) ? wpol : wref;
    const float* bs = (s < 2) ? bpol : bref;

    const int tid  = threadIdx.x;
    const int lane = tid & 31;
    const int wave = tid >> 5;
    const int half = lane >> 4;   // K-split half of the fragment
    const int idx  = lane & 15;   // row (A) / column (B,C) index

    const int row0 = rt * MTILE;

    // ---- stage x tile (32 x 2048) as bf16 in dynamic LDS ------------------
    extern __shared__ __align__(16) unsigned char smem_raw[];
    __bf16* lds_x = (__bf16*)smem_raw;                 // 32*2048*2 = 128 KB
    for (int e = tid; e < MTILE * HD; e += 256) {
        int r = e >> 11, k = e & (HD - 1);
        lds_x[e] = f32_to_bf16(xs[(size_t)(row0 + r) * HD + k]);
    }
    __syncthreads();

    float m_run[2][8], s_run[2][8];
#pragma unroll
    for (int rb = 0; rb < 2; ++rb)
#pragma unroll
        for (int v = 0; v < 8; ++v) { m_run[rb][v] = NEG_INF; s_run[rb][v] = 0.f; }

    const int nbase = chunk * NCHUNK;
    const __bf16* arow0 = lds_x + idx * HD + 16 * half;         // rows 0..15
    const __bf16* arow1 = lds_x + (16 + idx) * HD + 16 * half;  // rows 16..31

    for (int it = 0; it < NITERS; ++it) {
        const int n0 = nbase + it * (NWAVES * 32) + wave * 32;  // 32 cols/wave
        v8f c00 = {}, c01 = {}, c10 = {}, c11 = {};

        const size_t wrow0 = (size_t)(n0 + idx) * HD + 16 * half;
        const size_t wrow1 = (size_t)(n0 + 16 + idx) * HD + 16 * half;
        const float*  wf0 = BF16W ? nullptr : ((const float*)W + wrow0);
        const float*  wf1 = BF16W ? nullptr : ((const float*)W + wrow1);
        const __bf16* wb0 = BF16W ? ((const __bf16*)W + wrow0) : nullptr;
        const __bf16* wb1 = BF16W ? ((const __bf16*)W + wrow1) : nullptr;

        if (it + 1 < NITERS) {   // prefetch next iter's weight rows
            const size_t adv = (size_t)(NWAVES * 32) * HD;
            if (BF16W) {
                __builtin_prefetch((const __bf16*)W + wrow0 + adv, 0, 3);
                __builtin_prefetch((const __bf16*)W + wrow1 + adv, 0, 3);
            } else {
                __builtin_prefetch((const float*)W + wrow0 + adv, 0, 3);
                __builtin_prefetch((const float*)W + wrow1 + adv, 0, 3);
            }
        }

#pragma unroll 2
        for (int k0 = 0; k0 < HD; k0 += 32) {
            // A fragments: row=idx (+16), K = k0 + 16h + (0..15) contiguous
            v8bf a0lo = *(const v8bf*)(arow0 + k0);
            v8bf a0hi = *(const v8bf*)(arow0 + k0 + 8);
            v8bf a1lo = *(const v8bf*)(arow1 + k0);
            v8bf a1hi = *(const v8bf*)(arow1 + k0 + 8);
            v16bf a0 = __builtin_shufflevector(a0lo, a0hi,
                        0,1,2,3,4,5,6,7,8,9,10,11,12,13,14,15);
            v16bf a1 = __builtin_shufflevector(a1lo, a1hi,
                        0,1,2,3,4,5,6,7,8,9,10,11,12,13,14,15);
            // B fragments: col=idx (+16), K = k0 + 16h + (0..15) contiguous
            v16bf b0, b1;
            if (BF16W) {
                b0 = *(const v16bf*)(wb0 + k0);
                b1 = *(const v16bf*)(wb1 + k0);
            } else {
                float4 q0 = *(const float4*)(wf0 + k0);
                float4 q1 = *(const float4*)(wf0 + k0 + 4);
                float4 q2 = *(const float4*)(wf0 + k0 + 8);
                float4 q3 = *(const float4*)(wf0 + k0 + 12);
                b0[0]=f32_to_bf16(q0.x); b0[1]=f32_to_bf16(q0.y); b0[2]=f32_to_bf16(q0.z); b0[3]=f32_to_bf16(q0.w);
                b0[4]=f32_to_bf16(q1.x); b0[5]=f32_to_bf16(q1.y); b0[6]=f32_to_bf16(q1.z); b0[7]=f32_to_bf16(q1.w);
                b0[8]=f32_to_bf16(q2.x); b0[9]=f32_to_bf16(q2.y); b0[10]=f32_to_bf16(q2.z); b0[11]=f32_to_bf16(q2.w);
                b0[12]=f32_to_bf16(q3.x); b0[13]=f32_to_bf16(q3.y); b0[14]=f32_to_bf16(q3.z); b0[15]=f32_to_bf16(q3.w);
                q0 = *(const float4*)(wf1 + k0);
                q1 = *(const float4*)(wf1 + k0 + 4);
                q2 = *(const float4*)(wf1 + k0 + 8);
                q3 = *(const float4*)(wf1 + k0 + 12);
                b1[0]=f32_to_bf16(q0.x); b1[1]=f32_to_bf16(q0.y); b1[2]=f32_to_bf16(q0.z); b1[3]=f32_to_bf16(q0.w);
                b1[4]=f32_to_bf16(q1.x); b1[5]=f32_to_bf16(q1.y); b1[6]=f32_to_bf16(q1.z); b1[7]=f32_to_bf16(q1.w);
                b1[8]=f32_to_bf16(q2.x); b1[9]=f32_to_bf16(q2.y); b1[10]=f32_to_bf16(q2.z); b1[11]=f32_to_bf16(q2.w);
                b1[12]=f32_to_bf16(q3.x); b1[13]=f32_to_bf16(q3.y); b1[14]=f32_to_bf16(q3.z); b1[15]=f32_to_bf16(q3.w);
            }
            c00 = __builtin_amdgcn_wmma_f32_16x16x32_bf16(false, a0, false, b0, (short)0, c00, false, false);
            c10 = __builtin_amdgcn_wmma_f32_16x16x32_bf16(false, a1, false, b0, (short)0, c10, false, false);
            c01 = __builtin_amdgcn_wmma_f32_16x16x32_bf16(false, a0, false, b1, (short)0, c01, false, false);
            c11 = __builtin_amdgcn_wmma_f32_16x16x32_bf16(false, a1, false, b1, (short)0, c11, false, false);
        }

        const float bn0 = bs[n0 + idx];
        const float bn1 = bs[n0 + 16 + idx];
        fold_tile(c00, bn0, m_run[0], s_run[0]);
        fold_tile(c01, bn1, m_run[0], s_run[0]);
        fold_tile(c10, bn0, m_run[1], s_run[1]);
        fold_tile(c11, bn1, m_run[1], s_run[1]);
    }

    // ---- cross-wave merge (alias the x tile, safe after barrier) ----------
    __syncthreads();
    float* lm = (float*)smem_raw;          // [8 waves][32 rows]
    float* ls = lm + NWAVES * MTILE;
    if (idx == 0) {
#pragma unroll
        for (int rb = 0; rb < 2; ++rb)
#pragma unroll
            for (int v = 0; v < 8; ++v) {
                lm[wave * MTILE + rb * 16 + v + 8 * half] = m_run[rb][v];
                ls[wave * MTILE + rb * 16 + v + 8 * half] = s_run[rb][v];
            }
    }
    __syncthreads();
    if (tid < MTILE) {
        float M = NEG_INF;
#pragma unroll
        for (int w = 0; w < NWAVES; ++w) M = fmaxf(M, lm[w * MTILE + tid]);
        float S = 0.f;
#pragma unroll
        for (int w = 0; w < NWAVES; ++w) S += ls[w * MTILE + tid] * __expf(lm[w * MTILE + tid] - M);
        size_t gr = (size_t)s * ROWS + row0 + tid;
        pm[gr * NCHUNKS + chunk] = M;
        ps[gr * NCHUNKS + chunk] = S;
    }
}

// ---------------------------------------------------------------------------
// Kernel 3: target logit per token (one wave per token, f32 dot over H)
// ---------------------------------------------------------------------------
__global__ __launch_bounds__(256)
void dpo_target_logit_kernel(const float* __restrict__ x0, const float* __restrict__ x1,
                             const float* __restrict__ x2, const float* __restrict__ x3,
                             const float* __restrict__ wpol, const float* __restrict__ wref,
                             const float* __restrict__ bpol, const float* __restrict__ bref,
                             const int* __restrict__ tcho, const int* __restrict__ trej,
                             float* __restrict__ tl) {
    const int gw   = (int)((blockIdx.x * blockDim.x + threadIdx.x) >> 5);
    const int lane = threadIdx.x & 31;
    if (gw >= NSTREAMS * ROWS) return;
    const int s = gw >> 10, row = gw & (ROWS - 1);

    const int t = ((s & 1) ? trej : tcho)[row];
    float r = 0.f;
    if (t != IGNORE_IDX) {
        const float* xs = (s == 0) ? x0 : (s == 1) ? x1 : (s == 2) ? x2 : x3;
        const float* W  = (s < 2) ? wpol : wref;
        const float* b  = (s < 2) ? bpol : bref;
        const float* xr = xs + (size_t)row * HD;
        const float* wr = W + (size_t)t * HD;
        for (int k = lane; k < HD; k += 32) r = fmaf(xr[k], wr[k], r);
        r += __shfl_xor(r, 16, 32);
        r += __shfl_xor(r, 8, 32);
        r += __shfl_xor(r, 4, 32);
        r += __shfl_xor(r, 2, 32);
        r += __shfl_xor(r, 1, 32);
        r += b[t];
    }
    if (lane == 0) tl[gw] = r;
}

// ---------------------------------------------------------------------------
// Kernel 4: merge partial lse, per-sequence logp sums, DPO scalars
// ---------------------------------------------------------------------------
__global__ __launch_bounds__(256)
void dpo_combine_kernel(const float* __restrict__ pm, const float* __restrict__ ps,
                        const float* __restrict__ tl,
                        const int* __restrict__ tcho, const int* __restrict__ trej,
                        float* __restrict__ out) {
    __shared__ float seq[8];               // [stream][b]
    if (threadIdx.x < 8) seq[threadIdx.x] = 0.f;
    __syncthreads();

    for (int e = threadIdx.x; e < NSTREAMS * ROWS; e += 256) {
        const int s = e >> 10, row = e & (ROWS - 1);
        const int t = ((s & 1) ? trej : tcho)[row];
        if (t == IGNORE_IDX) continue;
        const size_t base = (size_t)e * NCHUNKS;
        float M = NEG_INF;
        for (int c = 0; c < NCHUNKS; ++c) M = fmaxf(M, pm[base + c]);
        float S = 0.f;
        for (int c = 0; c < NCHUNKS; ++c) S += ps[base + c] * __expf(pm[base + c] - M);
        const float lse = M + __logf(S);
        atomicAdd(&seq[s * 2 + (row >> 9)], tl[e] - lse);   // b = row / 512
    }
    __syncthreads();

    if (threadIdx.x == 0) {
        float loss = 0.f, crm = 0.f, rrm = 0.f, margin = 0.f, acc = 0.f;
#pragma unroll
        for (int b = 0; b < 2; ++b) {
            const float c  = seq[0 * 2 + b], r  = seq[1 * 2 + b];
            const float rc = seq[2 * 2 + b], rr = seq[3 * 2 + b];
            const float diff = BETA_C * ((c - r) - (rc - rr));
            const float lsg  = fminf(diff, 0.f) - log1pf(__expf(-fabsf(diff)));
            loss += -lsg;
            const float cw = BETA_C * (c - rc);
            const float rw = BETA_C * (r - rr);
            crm += cw; rrm += rw; margin += cw - rw; acc += (cw > rw) ? 1.f : 0.f;
        }
        out[0] = loss * 0.5f;
        out[1] = crm * 0.5f;
        out[2] = rrm * 0.5f;
        out[3] = margin * 0.5f;
        out[4] = acc * 0.5f;
        out[5] = 0.f;
    }
}

// ---------------------------------------------------------------------------
extern "C" void kernel_launch(void* const* d_in, const int* in_sizes, int n_in,
                              void* d_out, int out_size, void* d_ws, size_t ws_size,
                              hipStream_t stream) {
    (void)in_sizes; (void)n_in; (void)out_size;
    const float* x0  = (const float*)d_in[0];   // x_chosen
    const float* x1  = (const float*)d_in[1];   // x_rejected
    const float* x2  = (const float*)d_in[2];   // ref_x_chosen
    const float* x3  = (const float*)d_in[3];   // ref_x_rejected
    const float* wp  = (const float*)d_in[4];   // weight (V,H)
    const float* bp  = (const float*)d_in[5];   // bias (V)
    const float* wr  = (const float*)d_in[6];   // ref_weight
    const float* br  = (const float*)d_in[7];   // ref_bias
    const int*   tc  = (const int*)d_in[8];     // target_chosen
    const int*   tr  = (const int*)d_in[9];     // target_rejected
    float* out = (float*)d_out;

    const size_t welts = (size_t)VOC * HD;
    const size_t pelts = (size_t)NSTREAMS * ROWS * NCHUNKS;
    const size_t need_bf16 = 2 * welts * 2 + (2 * pelts + NSTREAMS * ROWS) * 4;
    const bool use_bf16w = ws_size >= need_bf16;

    unsigned short* wbf = (unsigned short*)d_ws;
    float* pm = use_bf16w ? (float*)((char*)d_ws + 2 * welts * 2) : (float*)d_ws;
    float* ps = pm + pelts;
    float* tl = ps + pelts;

    const unsigned gemm_grid = NSTREAMS * ROWTILES * NCHUNKS;   // 3200
    const unsigned lds_bytes = MTILE * HD * sizeof(unsigned short);  // 128 KB

    if (use_bf16w) {
        const size_t n4 = (2 * welts) / 4;
        dpo_convert_w_kernel<<<dim3((unsigned)((n4 + 255) / 256)), dim3(256), 0, stream>>>(
            wp, wr, wbf);
        dpo_gemm_lse_kernel<true><<<dim3(gemm_grid), dim3(256), lds_bytes, stream>>>(
            x0, x1, x2, x3, (const void*)wbf, (const void*)(wbf + welts), bp, br, pm, ps);
    } else {
        dpo_gemm_lse_kernel<false><<<dim3(gemm_grid), dim3(256), lds_bytes, stream>>>(
            x0, x1, x2, x3, (const void*)wp, (const void*)wr, bp, br, pm, ps);
    }
    dpo_target_logit_kernel<<<dim3((NSTREAMS * ROWS * 32) / 256), dim3(256), 0, stream>>>(
        x0, x1, x2, x3, wp, wr, bp, br, tc, tr, tl);
    dpo_combine_kernel<<<dim3(1), dim3(256), 0, stream>>>(pm, ps, tl, tc, tr, out);
}
